// SelfAttention_17781164605802
// MI455X (gfx1250) — matrix operand order repaired and checked by reading, not compiled, not run on hardware
//
#include <hip/hip_runtime.h>
#include <hip/hip_bf16.h>

typedef __bf16 bf16;
typedef __attribute__((ext_vector_type(16))) __bf16 v16bf;
typedef __attribute__((ext_vector_type(8)))  __bf16 v8bf;
typedef __attribute__((ext_vector_type(8)))  float  v8f;

static constexpr int Bn = 8, T = 2048, E = 1024, H = 64;

// Native bf16 conversions: let clang lower to packed cvt ops (RNE by default).
__device__ __forceinline__ bf16  f2bf(float f) { return (bf16)f; }
__device__ __forceinline__ float bf2f(bf16 b)  { return (float)b; }
__device__ __forceinline__ v8bf cvt8(v8f f)    { return __builtin_convertvector(f, v8bf); }

// ---------------- Kernel 0: W (E x H, f32) -> W^T (H x E, bf16), x3 ----------------
__global__ void wt_convert(const float* __restrict__ Wq, const float* __restrict__ Wk,
                           const float* __restrict__ Wv,
                           bf16* __restrict__ wtq, bf16* __restrict__ wtk,
                           bf16* __restrict__ wtv) {
  int idx   = blockIdx.x * blockDim.x + threadIdx.x;   // 0 .. 3*E*H-1
  int which = idx / (E * H);
  int r     = idx % (E * H);
  int h = r / E, ec = r % E;
  const float* src = (which == 0) ? Wq : ((which == 1) ? Wk : Wv);
  bf16*        dst = (which == 0) ? wtq : ((which == 1) ? wtk : wtv);
  dst[(size_t)h * E + ec] = f2bf(src[(size_t)ec * H + h]);
}

// ---------------- Kernel 1: QKV projection, shared-A triple WMMA ----------------
// q,k stored bf16 [B*T][H]; v stored transposed bf16 [B][H][T]
__global__ void qkv_kernel(const float* __restrict__ x,
                           const bf16* __restrict__ wtq, const bf16* __restrict__ wtk,
                           const bf16* __restrict__ wtv,
                           bf16* __restrict__ qb, bf16* __restrict__ kb,
                           bf16* __restrict__ vt) {
  const int lane = threadIdx.x & 31;
  const int wave = threadIdx.x >> 5;     // n-tile 0..3
  const int m0   = blockIdx.x * 16;      // row block over B*T
  const int col  = lane & 15;
  const int half = lane >> 4;
  const int n    = wave * 16 + col;      // 0..63

  v8f accq = {}; v8f acck = {}; v8f accv = {};

  const float* xrow = x + (size_t)(m0 + col) * E;
  const bf16*  bq   = wtq + (size_t)n * E;
  const bf16*  bk   = wtk + (size_t)n * E;
  const bf16*  bv   = wtv + (size_t)n * E;

  for (int kk = 0; kk < E; kk += 32) {
    // x is streamed exactly once: non-temporal so it doesn't evict the W^T buffers from L2.
    v8f f0 = __builtin_nontemporal_load((const v8f*)(xrow + kk + half * 16));
    v8f f1 = __builtin_nontemporal_load((const v8f*)(xrow + kk + half * 16 + 8));
    v8bf c0 = cvt8(f0), c1 = cvt8(f1);
    v16bf a;
#pragma unroll
    for (int j = 0; j < 8; ++j) { a[j] = c0[j]; a[8 + j] = c1[j]; }

    const int ko = kk + half * 16;
    v8bf q0 = *(const v8bf*)(bq + ko); v8bf q1 = *(const v8bf*)(bq + ko + 8);
    v8bf k0 = *(const v8bf*)(bk + ko); v8bf k1 = *(const v8bf*)(bk + ko + 8);
    v8bf v0 = *(const v8bf*)(bv + ko); v8bf v1 = *(const v8bf*)(bv + ko + 8);
    v16bf bbq, bbk, bbv;
#pragma unroll
    for (int j = 0; j < 8; ++j) {
      bbq[j] = q0[j]; bbq[8 + j] = q1[j];
      bbk[j] = k0[j]; bbk[8 + j] = k1[j];
      bbv[j] = v0[j]; bbv[8 + j] = v1[j];
    }
    accq = __builtin_amdgcn_wmma_f32_16x16x32_bf16(false, a, false, bbq, (short)0, accq, false, false);
    acck = __builtin_amdgcn_wmma_f32_16x16x32_bf16(false, a, false, bbk, (short)0, acck, false, false);
    accv = __builtin_amdgcn_wmma_f32_16x16x32_bf16(false, a, false, bbv, (short)0, accv, false, false);
  }
#pragma unroll
  for (int r = 0; r < 8; ++r) {
    int mg = m0 + r + 8 * half;               // global row in B*T
    qb[(size_t)mg * H + n] = f2bf(accq[r]);
    kb[(size_t)mg * H + n] = f2bf(acck[r]);
    int b = mg / T, t = mg % T;
    vt[((size_t)b * H + n) * T + t] = f2bf(accv[r]);  // v transposed
  }
}

// ---------------- Kernel 2: e = exp(q k^T / 8), bf16 [B][T][T] ----------------
__global__ void scores_kernel(const bf16* __restrict__ qb, const bf16* __restrict__ kb,
                              bf16* __restrict__ e) {
  const int lane = threadIdx.x & 31;
  const int wave = threadIdx.x >> 5;
  const int tm = blockIdx.x;                  // 0..127
  const int tn = blockIdx.y * 4 + wave;       // 0..127
  const int b  = blockIdx.z;
  const int col = lane & 15, half = lane >> 4;

  const bf16* arow = qb + ((size_t)b * T + tm * 16 + col) * H;
  const bf16* brow = kb + ((size_t)b * T + tn * 16 + col) * H;  // B = k^T -> row of k

  v8f acc = {};
#pragma unroll
  for (int kk = 0; kk < H; kk += 32) {
    v8bf a0 = *(const v8bf*)(arow + kk + half * 16);
    v8bf a1 = *(const v8bf*)(arow + kk + half * 16 + 8);
    v8bf b0 = *(const v8bf*)(brow + kk + half * 16);
    v8bf b1 = *(const v8bf*)(brow + kk + half * 16 + 8);
    v16bf av, bv;
#pragma unroll
    for (int j = 0; j < 8; ++j) { av[j] = a0[j]; av[8 + j] = a1[j];
                                  bv[j] = b0[j]; bv[8 + j] = b1[j]; }
    acc = __builtin_amdgcn_wmma_f32_16x16x32_bf16(false, av, false, bv, (short)0, acc, false, false);
  }
#pragma unroll
  for (int r = 0; r < 8; ++r) {
    int trow = tm * 16 + r + 8 * half;
    int scol = tn * 16 + col;
    e[((size_t)b * T + trow) * T + scol] = f2bf(__expf(acc[r] * 0.125f));
  }
}

// ---------------- Kernel 3: softmax over batch axis, in place on e ----------------
__global__ void bsoftmax_kernel(bf16* __restrict__ e) {
  int gid  = blockIdx.x * blockDim.x + threadIdx.x;  // T*T/8 threads
  int t    = gid >> 8;                                // / (T/8)
  int soff = (gid & 255) << 3;                        // 8 s-positions per thread
  v8f rows[Bn];
  v8f sum = {};
#pragma unroll
  for (int bb = 0; bb < Bn; ++bb) {
    v8bf rv = *(const v8bf*)(e + ((size_t)bb * T + t) * T + soff);
    rows[bb] = __builtin_convertvector(rv, v8f);
    sum += rows[bb];
  }
  v8f inv;
#pragma unroll
  for (int j = 0; j < 8; ++j) inv[j] = 1.0f / sum[j];
#pragma unroll
  for (int bb = 0; bb < Bn; ++bb) {
    v8bf o = cvt8(rows[bb] * inv);
    *(v8bf*)(e + ((size_t)bb * T + t) * T + soff) = o;
  }
}

// ---------------- Kernel 4: out = p v, fp32 [B][T][H] ----------------
__global__ void pv_kernel(const bf16* __restrict__ e, const bf16* __restrict__ vt,
                          float* __restrict__ out) {
  const int lane = threadIdx.x & 31;
  const int wave = threadIdx.x >> 5;     // n-tile 0..3
  const int tm = blockIdx.x;             // 0..127
  const int b  = blockIdx.y;
  const int col = lane & 15, half = lane >> 4;
  const int n = wave * 16 + col;

  const bf16* arow = e  + ((size_t)b * T + tm * 16 + col) * T;
  const bf16* brow = vt + ((size_t)b * H + n) * T;   // contiguous thanks to v^T

  v8f acc = {};
  for (int kk = 0; kk < T; kk += 32) {
    // prefetch next k-step (speculative; dropped if past end)
    __builtin_prefetch(arow + kk + 32, 0, 1);
    __builtin_prefetch(brow + kk + 32, 0, 1);
    v8bf a0 = *(const v8bf*)(arow + kk + half * 16);
    v8bf a1 = *(const v8bf*)(arow + kk + half * 16 + 8);
    v8bf b0 = *(const v8bf*)(brow + kk + half * 16);
    v8bf b1 = *(const v8bf*)(brow + kk + half * 16 + 8);
    v16bf av, bv;
#pragma unroll
    for (int j = 0; j < 8; ++j) { av[j] = a0[j]; av[8 + j] = a1[j];
                                  bv[j] = b0[j]; bv[8 + j] = b1[j]; }
    acc = __builtin_amdgcn_wmma_f32_16x16x32_bf16(false, av, false, bv, (short)0, acc, false, false);
  }
#pragma unroll
  for (int r = 0; r < 8; ++r) {
    int trow = tm * 16 + r + 8 * half;
    out[((size_t)b * T + trow) * H + n] = acc[r];
  }
}

extern "C" void kernel_launch(void* const* d_in, const int* in_sizes, int n_in,
                              void* d_out, int out_size, void* d_ws, size_t ws_size,
                              hipStream_t stream) {
  (void)in_sizes; (void)n_in; (void)out_size; (void)ws_size;
  const float* x  = (const float*)d_in[0];
  const float* Wq = (const float*)d_in[1];
  const float* Wk = (const float*)d_in[2];
  const float* Wv = (const float*)d_in[3];
  float* out = (float*)d_out;

  // workspace carve-up (bf16 elements): 3*H*E + 3*B*T*H + B*T*T  (~70.5 MB)
  bf16* p   = (bf16*)d_ws;
  bf16* wtq = p; p += (size_t)H * E;
  bf16* wtk = p; p += (size_t)H * E;
  bf16* wtv = p; p += (size_t)H * E;
  bf16* qb  = p; p += (size_t)Bn * T * H;
  bf16* kb  = p; p += (size_t)Bn * T * H;
  bf16* vt  = p; p += (size_t)Bn * T * H;
  bf16* e   = p;                      // Bn*T*T

  wt_convert<<<(3 * E * H) / 256, 256, 0, stream>>>(Wq, Wk, Wv, wtq, wtk, wtv);
  qkv_kernel<<<(Bn * T) / 16, 128, 0, stream>>>(x, wtq, wtk, wtv, qb, kb, vt);
  scores_kernel<<<dim3(T / 16, T / 64, Bn), 128, 0, stream>>>(qb, kb, e);
  bsoftmax_kernel<<<(T / 8) * (T / 256), 256, 0, stream>>>(e);
  pv_kernel<<<dim3(T / 16, Bn), 128, 0, stream>>>(e, vt, out);
}
